// MoELayer_53781580480968
// MI455X (gfx1250) — matrix operand order repaired and checked by reading, not compiled, not run on hardware
//
#include <hip/hip_runtime.h>
#include <math.h>

// ---------------------------------------------------------------------------
// MoE top-2 FFN for gfx1250 (MI455X), f16 WMMA path.
//   x:[4,2048,1024] f32, gate_w:[1024,8], w1:[8,1024,4096], w2:[8,4096,1024]
// Weights are pre-transposed + converted to f16 once (w1t:[e][H][C],
// w2t:[e][C][H]) so both GEMM A- and B-tile staging are K-contiguous
// b128 copies; all LDS traffic is ds_load_b128/ds_store_b128.
// Workspace (~294 MB): w1t | w2t | xh | hws | routing arrays.
// ---------------------------------------------------------------------------

typedef __attribute__((ext_vector_type(16))) _Float16 v16h;
typedef __attribute__((ext_vector_type(8)))  _Float16 v8h;
typedef __attribute__((ext_vector_type(4)))  _Float16 v4h;
typedef __attribute__((ext_vector_type(8)))  float    v8f;

union V16 { v16h v; v8h h[2]; };

#define NTOK   8192
#define CDIM   1024
#define HDIM   4096
#define NEXP   8
#define MAXT   136        // max m-tiles: 16384/128 + 8 pad tiles
#define MAXR   17408      // max padded routed rows

// ------ kernel 1: f32 -> f16 conversion with 32x32 tiled transpose ---------
// src: per-expert [R][S] f32 ; dst: per-expert [S][R] f16 ; expert = blockIdx.z
__global__ __launch_bounds__(256) void transpose_cvt_kernel(
    const float* __restrict__ src, _Float16* __restrict__ dst, int R, int S)
{
    __shared__ float t[32][33];
    const float* se = src + (size_t)blockIdx.z * R * S;
    _Float16*    de = dst + (size_t)blockIdx.z * R * S;
    const int r0 = blockIdx.y * 32, s0 = blockIdx.x * 32;
    const int tr = threadIdx.x >> 3, tc = (threadIdx.x & 7) * 4;
    float4 v = *(const float4*)(se + (size_t)(r0 + tr) * S + s0 + tc);
    t[tr][tc+0] = v.x; t[tr][tc+1] = v.y; t[tr][tc+2] = v.z; t[tr][tc+3] = v.w;
    __syncthreads();
    v4h o;
    #pragma unroll
    for (int j = 0; j < 4; ++j) o[j] = (_Float16)t[tc + j][tr];
    *(v4h*)(de + (size_t)(s0 + tr) * R + r0 + tc) = o;
}

// ---------------- kernel 0: zero expert counters ---------------------------
__global__ void zero_kernel(int* counts) {
    if (threadIdx.x < NEXP) counts[threadIdx.x] = 0;
}

// ---------------- kernel 2: gating (one wave32 per token) ------------------
__global__ __launch_bounds__(128) void gate_kernel(
    const float* __restrict__ x, const float* __restrict__ gw,
    const float* __restrict__ gb, _Float16* __restrict__ xh,
    int* __restrict__ topi, float* __restrict__ topw, int* __restrict__ counts)
{
    const int wave = threadIdx.x >> 5, lane = threadIdx.x & 31;
    const int t = blockIdx.x * 4 + wave;
    const float* xr = x + (size_t)t * CDIM;
    float acc[NEXP];
    #pragma unroll
    for (int e = 0; e < NEXP; ++e) acc[e] = 0.f;
    for (int i = 0; i < CDIM / 32; ++i) {
        int k = i * 32 + lane;
        float xv = xr[k];
        xh[(size_t)t * CDIM + k] = (_Float16)xv;
        const float* g = gw + k * NEXP;
        #pragma unroll
        for (int e = 0; e < NEXP; ++e) acc[e] += xv * g[e];
    }
    #pragma unroll
    for (int e = 0; e < NEXP; ++e)
        for (int off = 16; off; off >>= 1)
            acc[e] += __shfl_xor(acc[e], off, 32);
    if (lane == 0) {
        float lg[NEXP];
        #pragma unroll
        for (int e = 0; e < NEXP; ++e) lg[e] = acc[e] + gb[e];
        int e0 = 0;
        for (int e = 1; e < NEXP; ++e) if (lg[e] > lg[e0]) e0 = e;
        int e1 = -1;
        for (int e = 0; e < NEXP; ++e)
            if (e != e0 && (e1 < 0 || lg[e] > lg[e1])) e1 = e;
        float z  = expf(lg[e1] - lg[e0]);     // <= 1, stable
        float p0 = 1.f / (1.f + z);
        topi[t*2] = e0; topi[t*2+1] = e1;
        topw[t*2] = p0; topw[t*2+1] = 1.f - p0;
        atomicAdd(&counts[e0], 1);
        atomicAdd(&counts[e1], 1);
    }
}

// ---------------- kernel 3: routing prep (single block) --------------------
__global__ __launch_bounds__(256) void prep_kernel(
    const int* __restrict__ counts, int* __restrict__ fill,
    int* __restrict__ tile_expert, int* __restrict__ tile_rowbase,
    int* __restrict__ rows_l, float* __restrict__ roww)
{
    for (int i = threadIdx.x; i < MAXR; i += blockDim.x) {
        rows_l[i] = -1; roww[i] = 0.f;
    }
    if (threadIdx.x == 0) {
        int total = 0, t = 0;
        for (int e = 0; e < NEXP; ++e) {
            fill[e] = total;
            int nt = (counts[e] + 127) / 128;
            for (int i = 0; i < nt && t < MAXT; ++i, ++t) {
                tile_expert[t] = e; tile_rowbase[t] = total + i * 128;
            }
            total += nt * 128;
        }
        for (; t < MAXT; ++t) { tile_expert[t] = -1; tile_rowbase[t] = 0; }
    }
}

// ---------------- kernel 4: scatter tokens into expert row lists -----------
__global__ __launch_bounds__(256) void scatter_kernel(
    const int* __restrict__ topi, const float* __restrict__ topw,
    int* __restrict__ fill, int* __restrict__ rows_l, float* __restrict__ roww)
{
    int t = blockIdx.x * 256 + threadIdx.x;
    if (t >= NTOK) return;
    #pragma unroll
    for (int k = 0; k < 2; ++k) {
        int e = topi[t*2+k];
        int pos = atomicAdd(&fill[e], 1);
        rows_l[pos] = t;
        roww[pos] = topw[t*2+k];
    }
}

// ---------------- kernel 5: out = sum_i p_i * b2[e_i] ----------------------
__global__ __launch_bounds__(256) void outinit_kernel(
    const int* __restrict__ topi, const float* __restrict__ topw,
    const float* __restrict__ b2, float* __restrict__ out)
{
    int t = blockIdx.x;
    int e0 = topi[t*2], e1 = topi[t*2+1];
    float p0 = topw[t*2], p1 = topw[t*2+1];
    const float4* b20 = (const float4*)(b2 + (size_t)e0 * CDIM);
    const float4* b21 = (const float4*)(b2 + (size_t)e1 * CDIM);
    float4 a = b20[threadIdx.x], b = b21[threadIdx.x], r;
    r.x = p0*a.x + p1*b.x; r.y = p0*a.y + p1*b.y;
    r.z = p0*a.z + p1*b.z; r.w = p0*a.w + p1*b.w;
    ((float4*)(out + (size_t)t * CDIM))[threadIdx.x] = r;
}

// ---------------- grouped GEMM kernels -------------------------------------
// 128x128 tile, BK=64, double-buffered LDS, register-staged pipeline.
// 8 waves: 4(M) x 2(N); wave tile 32x64 = 8 wmma frags (v_wmma_f32_16x16x32_f16)
#define BM 128
#define BN 128
#define BK 64
#define SA 72      // LDS row stride (halves): 144B, 16B-aligned, conflict-free
#define SB 72

// kernel 6: h = gelu(Xg @ W1 + b1), h -> f16 workspace
__global__ __launch_bounds__(256) void gemm1_kernel(
    const _Float16* __restrict__ xh, const _Float16* __restrict__ w1t,
    const float* __restrict__ b1, const int* __restrict__ rows_l,
    const int* __restrict__ tile_expert, const int* __restrict__ tile_rowbase,
    _Float16* __restrict__ hws)
{
    __shared__ __align__(16) _Float16 sA[2][BM * SA];
    __shared__ __align__(16) _Float16 sB[2][BN * SB];
    const int exp = tile_expert[blockIdx.x];
    if (exp < 0) return;
    const int mrow0 = tile_rowbase[blockIdx.x];
    const int n0 = blockIdx.y * BN;
    const float* b1e = b1 + (size_t)exp * HDIM;

    const int tid = threadIdx.x, lane = tid & 31, wave = tid >> 5;
    const int wm = wave & 3, wn = wave >> 2;

    // staging: 2 threads per row, each copies 32 halves (4 x b128)
    const int arow = tid >> 1, aseg = (tid & 1) * 32;
    const int token = rows_l[mrow0 + arow];
    const v8h* asrc = (token >= 0) ? (const v8h*)(xh + (size_t)token * CDIM)
                                   : nullptr;
    // w1t is [H][C] per expert: B rows (n) are K-contiguous
    const v8h* bsrc = (const v8h*)(w1t + (size_t)exp * CDIM * HDIM
                                       + (size_t)(n0 + arow) * CDIM);

    v8f acc[2][4];
    #pragma unroll
    for (int mi = 0; mi < 2; ++mi)
        #pragma unroll
        for (int ni = 0; ni < 4; ++ni) acc[mi][ni] = (v8f){};

    v8h ar[4], br[4];
    auto load_regs = [&](int k0) {
        #pragma unroll
        for (int i = 0; i < 4; ++i) {
            ar[i] = (token >= 0) ? asrc[(k0 + aseg) / 8 + i] : (v8h){};
            br[i] = bsrc[(k0 + aseg) / 8 + i];
        }
    };
    auto store_stage = [&](int buf) {
        #pragma unroll
        for (int i = 0; i < 4; ++i) {
            *(v8h*)&sA[buf][arow * SA + aseg + i * 8] = ar[i];
            *(v8h*)&sB[buf][arow * SB + aseg + i * 8] = br[i];
        }
    };
    auto compute = [&](int buf) {
        #pragma unroll
        for (int kk = 0; kk < BK; kk += 32) {
            V16 a[2], b[4];
            #pragma unroll
            for (int mi = 0; mi < 2; ++mi) {
                int r  = (wm * 32 + mi * 16 + (lane & 15)) * SA;
                int kb = kk + ((lane >> 4) << 3);       // ISA 16-bit A layout
                a[mi].h[0] = *(const v8h*)&sA[buf][r + kb];
                a[mi].h[1] = *(const v8h*)&sA[buf][r + kb + 16];
            }
            #pragma unroll
            for (int ni = 0; ni < 4; ++ni) {
                int c  = (wn * 64 + ni * 16 + (lane & 15)) * SB;
                int kb = kk + ((lane >> 4) << 3);       // ISA 16-bit A/B layout
                b[ni].h[0] = *(const v8h*)&sB[buf][c + kb];
                b[ni].h[1] = *(const v8h*)&sB[buf][c + kb + 16];
            }
            #pragma unroll
            for (int mi = 0; mi < 2; ++mi)
                #pragma unroll
                for (int ni = 0; ni < 4; ++ni)
                    acc[mi][ni] = __builtin_amdgcn_wmma_f32_16x16x32_f16(
                        false, a[mi].v, false, b[ni].v,
                        (short)0, acc[mi][ni], false, false);
        }
    };

    load_regs(0);
    store_stage(0);
    int cur = 0;
    const int nch = CDIM / BK;
    for (int c = 0; c < nch; ++c, cur ^= 1) {
        __syncthreads();
        if (c + 1 < nch) load_regs((c + 1) * BK);   // overlap with wmma below
        compute(cur);
        if (c + 1 < nch) store_stage(cur ^ 1);
    }
    // epilogue: + b1, exact GELU, -> f16 workspace
    #pragma unroll
    for (int mi = 0; mi < 2; ++mi) {
        int rbase = mrow0 + wm * 32 + mi * 16 + ((lane >> 4) << 3);
        #pragma unroll
        for (int ni = 0; ni < 4; ++ni) {
            int col = n0 + wn * 64 + ni * 16 + (lane & 15);
            float bb = b1e[col];
            #pragma unroll
            for (int g = 0; g < 8; ++g) {
                float v = acc[mi][ni][g] + bb;
                v = 0.5f * v * (1.0f + erff(v * 0.70710678118f));
                hws[(size_t)(rbase + g) * HDIM + col] = (_Float16)v;
            }
        }
    }
}

// kernel 7: out += p * (h @ W2)
__global__ __launch_bounds__(256) void gemm2_kernel(
    const _Float16* __restrict__ hws, const _Float16* __restrict__ w2t,
    const int* __restrict__ rows_l, const float* __restrict__ roww,
    const int* __restrict__ tile_expert, const int* __restrict__ tile_rowbase,
    float* __restrict__ out)
{
    __shared__ __align__(16) _Float16 sA[2][BM * SA];
    __shared__ __align__(16) _Float16 sB[2][BN * SB];
    const int exp = tile_expert[blockIdx.x];
    if (exp < 0) return;
    const int mrow0 = tile_rowbase[blockIdx.x];
    const int n0 = blockIdx.y * BN;

    const int tid = threadIdx.x, lane = tid & 31, wave = tid >> 5;
    const int wm = wave & 3, wn = wave >> 2;
    const int arow = tid >> 1, aseg = (tid & 1) * 32;
    const v8h* asrc = (const v8h*)(hws + (size_t)(mrow0 + arow) * HDIM);
    // w2t is [C][H] per expert: B rows (n over C) are K-contiguous over H
    const v8h* bsrc = (const v8h*)(w2t + (size_t)exp * HDIM * CDIM
                                       + (size_t)(n0 + arow) * HDIM);

    v8f acc[2][4];
    #pragma unroll
    for (int mi = 0; mi < 2; ++mi)
        #pragma unroll
        for (int ni = 0; ni < 4; ++ni) acc[mi][ni] = (v8f){};

    v8h ar[4], br[4];
    auto load_regs = [&](int k0) {
        #pragma unroll
        for (int i = 0; i < 4; ++i) {
            ar[i] = asrc[(k0 + aseg) / 8 + i];
            br[i] = bsrc[(k0 + aseg) / 8 + i];
        }
    };
    auto store_stage = [&](int buf) {
        #pragma unroll
        for (int i = 0; i < 4; ++i) {
            *(v8h*)&sA[buf][arow * SA + aseg + i * 8] = ar[i];
            *(v8h*)&sB[buf][arow * SB + aseg + i * 8] = br[i];
        }
    };
    auto compute = [&](int buf) {
        #pragma unroll
        for (int kk = 0; kk < BK; kk += 32) {
            V16 a[2], b[4];
            #pragma unroll
            for (int mi = 0; mi < 2; ++mi) {
                int r  = (wm * 32 + mi * 16 + (lane & 15)) * SA;
                int kb = kk + ((lane >> 4) << 3);
                a[mi].h[0] = *(const v8h*)&sA[buf][r + kb];
                a[mi].h[1] = *(const v8h*)&sA[buf][r + kb + 16];
            }
            #pragma unroll
            for (int ni = 0; ni < 4; ++ni) {
                int c  = (wn * 64 + ni * 16 + (lane & 15)) * SB;
                int kb = kk + ((lane >> 4) << 3);
                b[ni].h[0] = *(const v8h*)&sB[buf][c + kb];
                b[ni].h[1] = *(const v8h*)&sB[buf][c + kb + 16];
            }
            #pragma unroll
            for (int mi = 0; mi < 2; ++mi)
                #pragma unroll
                for (int ni = 0; ni < 4; ++ni)
                    acc[mi][ni] = __builtin_amdgcn_wmma_f32_16x16x32_f16(
                        false, a[mi].v, false, b[ni].v,
                        (short)0, acc[mi][ni], false, false);
        }
    };

    load_regs(0);
    store_stage(0);
    int cur = 0;
    const int nch = HDIM / BK;
    for (int c = 0; c < nch; ++c, cur ^= 1) {
        __syncthreads();
        if (c + 1 < nch) load_regs((c + 1) * BK);
        compute(cur);
        if (c + 1 < nch) store_stage(cur ^ 1);
    }
    // epilogue: scale by routing weight, atomic-accumulate into out
    #pragma unroll
    for (int mi = 0; mi < 2; ++mi) {
        int rbase = mrow0 + wm * 32 + mi * 16 + ((lane >> 4) << 3);
        #pragma unroll
        for (int ni = 0; ni < 4; ++ni) {
            int col = n0 + wn * 64 + ni * 16 + (lane & 15);
            #pragma unroll
            for (int g = 0; g < 8; ++g) {
                int rowg = rbase + g;
                int tok  = rows_l[rowg];
                if (tok >= 0) {
                    float w = roww[rowg];
                    atomicAdd(&out[(size_t)tok * CDIM + col],
                              w * acc[mi][ni][g]);
                }
            }
        }
    }
}

// ---------------------------------------------------------------------------
extern "C" void kernel_launch(void* const* d_in, const int* in_sizes, int n_in,
                              void* d_out, int out_size, void* d_ws, size_t ws_size,
                              hipStream_t stream) {
    const float* x  = (const float*)d_in[0];
    const float* gw = (const float*)d_in[1];
    const float* gb = (const float*)d_in[2];
    const float* w1 = (const float*)d_in[3];
    const float* b1 = (const float*)d_in[4];
    const float* w2 = (const float*)d_in[5];
    const float* b2 = (const float*)d_in[6];
    float* out = (float*)d_out;

    // workspace carve-up (bytes)
    char* ws = (char*)d_ws;
    constexpr size_t W1N   = (size_t)NEXP * CDIM * HDIM;       // 33,554,432
    constexpr size_t OFF_W1T  = 0;
    constexpr size_t OFF_W2T  = OFF_W1T + W1N * 2;
    constexpr size_t OFF_XH   = OFF_W2T + W1N * 2;
    constexpr size_t OFF_HWS  = OFF_XH  + (size_t)NTOK * CDIM * 2;
    constexpr size_t OFF_ROWS = OFF_HWS + (size_t)MAXR * HDIM * 2;
    constexpr size_t OFF_ROWW = OFF_ROWS + (size_t)MAXR * 4;
    constexpr size_t OFF_TOPI = OFF_ROWW + (size_t)MAXR * 4;
    constexpr size_t OFF_TOPW = OFF_TOPI + (size_t)NTOK * 2 * 4;
    constexpr size_t OFF_CNT  = OFF_TOPW + (size_t)NTOK * 2 * 4;
    constexpr size_t OFF_FILL = OFF_CNT  + 256;
    constexpr size_t OFF_TEXP = OFF_FILL + 256;
    constexpr size_t OFF_TROW = OFF_TEXP + 1024;

    _Float16* w1t  = (_Float16*)(ws + OFF_W1T);
    _Float16* w2t  = (_Float16*)(ws + OFF_W2T);
    _Float16* xh   = (_Float16*)(ws + OFF_XH);
    _Float16* hws  = (_Float16*)(ws + OFF_HWS);
    int*    rows_l = (int*)  (ws + OFF_ROWS);
    float*  roww   = (float*)(ws + OFF_ROWW);
    int*    topi   = (int*)  (ws + OFF_TOPI);
    float*  topw   = (float*)(ws + OFF_TOPW);
    int*    counts = (int*)  (ws + OFF_CNT);
    int*    fill   = (int*)  (ws + OFF_FILL);
    int*    texp   = (int*)  (ws + OFF_TEXP);
    int*    trow   = (int*)  (ws + OFF_TROW);

    // 0) zero expert counters
    zero_kernel<<<1, 32, 0, stream>>>(counts);
    // 1) weights f32 -> f16 with transpose (once): w1t:[e][H][C], w2t:[e][C][H]
    transpose_cvt_kernel<<<dim3(HDIM/32, CDIM/32, NEXP), 256, 0, stream>>>(
        w1, w1t, CDIM, HDIM);
    transpose_cvt_kernel<<<dim3(CDIM/32, HDIM/32, NEXP), 256, 0, stream>>>(
        w2, w2t, HDIM, CDIM);
    // 2) gating: logits, top-2, softmax; x -> f16
    gate_kernel<<<NTOK / 4, 128, 0, stream>>>(x, gw, gb, xh, topi, topw, counts);
    // 3) padded per-expert offsets + tile map; init row lists
    prep_kernel<<<1, 256, 0, stream>>>(counts, fill, texp, trow, rows_l, roww);
    // 4) scatter tokens into expert row lists
    scatter_kernel<<<NTOK / 256, 256, 0, stream>>>(topi, topw, fill, rows_l, roww);
    // 5) out = sum_i p_i * b2[e_i]
    outinit_kernel<<<NTOK, 256, 0, stream>>>(topi, topw, b2, out);
    // 6) h = gelu(Xg @ W1 + b1)  (grouped over experts)
    gemm1_kernel<<<dim3(MAXT, HDIM / BN), 256, 0, stream>>>(
        xh, w1t, b1, rows_l, texp, trow, hws);
    // 7) out += p * (h @ W2)
    gemm2_kernel<<<dim3(MAXT, CDIM / BN), 256, 0, stream>>>(
        hws, w2t, rows_l, roww, texp, trow, out);

    (void)in_sizes; (void)n_in; (void)out_size; (void)ws_size;
}
